// MaskRcnn_59777354825932
// MI455X (gfx1250) — compile-verified, matrix operand order checked by reading, not run on hardware
//
#include <hip/hip_runtime.h>
#include <math.h>

typedef __bf16 bf16_t;
typedef bf16_t v16bf __attribute__((ext_vector_type(16)));
typedef float  v8f   __attribute__((ext_vector_type(8)));

#define CDIV(a,b) (((a)+(b)-1)/(b))

struct alignas(16) BF8 { bf16_t h[8]; };

// branchless fragment gather: two contiguous 16-byte loads per lane
__device__ __forceinline__ v16bf frag_load(const bf16_t* __restrict__ row, int off0, int off1) {
  BF8 a = *reinterpret_cast<const BF8*>(row + off0);
  BF8 b = *reinterpret_cast<const BF8*>(row + off1);
  v16bf v;
#pragma unroll
  for (int e = 0; e < 8; ++e) { v[e] = a.h[e]; v[8 + e] = b.h[e]; }
  return v;
}

// ---------------- utility ----------------
__global__ void k_zero(unsigned int* p, long n) {
  long i = (long)blockIdx.x * 256 + threadIdx.x;
  if (i < n) p[i] = 0u;
}

__global__ void k_cvt(const float* __restrict__ w, bf16_t* __restrict__ o, long n) {
  long i = (long)blockIdx.x * 256 + threadIdx.x;
  if (i < n) o[i] = (bf16_t)w[i];
}

// o[(co*9+t)*Ci + ci] = w[(co*Ci+ci)*9 + t]   (conv weight -> [co][tap][ci] bf16)
__global__ void k_cvt_w3(const float* __restrict__ w, bf16_t* __restrict__ o, long n, int Ci) {
  long i = (long)blockIdx.x * 256 + threadIdx.x;
  if (i >= n) return;
  int ci = (int)(i % Ci);
  long r = i / Ci;
  int t = (int)(r % 9);
  long co = r / 9;
  o[i] = (bf16_t)w[(co * Ci + ci) * 9 + t];
}

// deconv weight (i,o,a,b) -> [ab][o][i] f32
__global__ void k_cvt_wdec(const float* __restrict__ w, float* __restrict__ o, long n) {
  long i = (long)blockIdx.x * 256 + threadIdx.x;
  if (i >= n) return;
  int ic = (int)(i % 256);
  long r = i / 256;
  int oc = (int)(r % 256);
  int ab = (int)(r / 256);
  o[i] = w[((long)ic * 256 + oc) * 4 + ab];
}

// CHW f32 -> padded (H+2)(W+2) NHWC bf16 (interior only; border pre-zeroed)
__global__ void k_padfeat(const float* __restrict__ f, bf16_t* __restrict__ pad, int H, int W) {
  long i = (long)blockIdx.x * 256 + threadIdx.x;
  long n = (long)H * W * 256;
  if (i >= n) return;
  int c = (int)(i % 256);
  long r = i / 256;
  int x = (int)(r % W);
  int y = (int)(r / W);
  pad[(((long)(y + 1) * (W + 2) + (x + 1)) * 256) + c] = (bf16_t)f[(long)c * H * W + (long)y * W + x];
}

// ---------------- WMMA implicit-GEMM 3x3 conv (pad=1) ----------------
// padin: [(H+2)*(W+2)*Cin] bf16 NHWC (per z: stride inStrideZ)
// bt:    [Cout][9][Cin] bf16, out: [Mvalid][Cout] f32 NHWC (per z: stride outStrideZ)
// zrow:  >= Cin zeroed bf16 elements; invalid M rows read from here (branchless)
__global__ void __launch_bounds__(32)
k_conv3_wmma(const bf16_t* __restrict__ padin, const bf16_t* __restrict__ bt,
             const float* __restrict__ bias, const bf16_t* __restrict__ zrow,
             float* __restrict__ out, int W, int Wp, int Cin, int Cout, int Mvalid, int relu,
             long inStrideZ, long outStrideZ)
{
  int lane = threadIdx.x;
  int hi = lane >> 4;
  int l15 = lane & 15;
  int m0 = blockIdx.x * 16;
  int n0 = blockIdx.y * 16;
  const bf16_t* pin = padin + (long)blockIdx.z * inStrideZ;
  float* pout = out + (long)blockIdx.z * outStrideZ;

  int mA = m0 + l15;
  bool av = (mA < Mvalid);
  int mc = av ? mA : 0;
  int y = mc / W;
  int x = mc - y * W;
  int nB = n0 + l15;

  v8f acc;
  float bz = bias ? bias[nB] : 0.0f;
#pragma unroll
  for (int e = 0; e < 8; ++e) acc[e] = bz;

  int o0 = hi * 8;
  int o1 = 16 + hi * 8;
  for (int t = 0; t < 9; ++t) {
    int ky = t / 3, kx = t - ky * 3;
    const bf16_t* arow = av ? (pin + ((long)(y + ky) * Wp + (x + kx)) * Cin) : zrow;
    const bf16_t* brow = bt + ((long)nB * 9 + t) * Cin;
    for (int c0 = 0; c0 < Cin; c0 += 32) {
      v16bf a = frag_load(arow, c0 + o0, c0 + o1);
      v16bf b = frag_load(brow, c0 + o0, c0 + o1);
      acc = __builtin_amdgcn_wmma_f32_16x16x32_bf16(false, a, false, b, (short)0, acc, false, false);
    }
  }
#pragma unroll
  for (int e = 0; e < 8; ++e) {
    int m = m0 + hi * 8 + e;
    if (m < Mvalid) {
      float v = acc[e];
      if (relu) v = fmaxf(v, 0.0f);
      pout[(long)m * Cout + (n0 + l15)] = v;
    }
  }
}

// ---------------- WMMA GEMM: out[M][N] = A[M][K] * Bt[N][K]^T + bias ----------------
__global__ void __launch_bounds__(32)
k_gemm_wmma(const bf16_t* __restrict__ A, const bf16_t* __restrict__ Bt,
            const float* __restrict__ bias, float* __restrict__ out,
            int K, int N, int relu)
{
  int lane = threadIdx.x;
  int hi = lane >> 4;
  int l15 = lane & 15;
  int m0 = blockIdx.x * 16;
  int n0 = blockIdx.y * 16;
  const bf16_t* arow = A + (long)(m0 + l15) * K;
  const bf16_t* brow = Bt + (long)(n0 + l15) * K;

  v8f acc;
  float bz = bias ? bias[n0 + l15] : 0.0f;
#pragma unroll
  for (int e = 0; e < 8; ++e) acc[e] = bz;

  int o0 = hi * 8;
  int o1 = 16 + hi * 8;
  for (int k0 = 0; k0 < K; k0 += 32) {
    v16bf a = frag_load(arow, k0 + o0, k0 + o1);
    v16bf b = frag_load(brow, k0 + o0, k0 + o1);
    acc = __builtin_amdgcn_wmma_f32_16x16x32_bf16(false, a, false, b, (short)0, acc, false, false);
  }
#pragma unroll
  for (int e = 0; e < 8; ++e) {
    int m = m0 + hi * 8 + e;
    float v = acc[e];
    if (relu) v = fmaxf(v, 0.0f);
    out[(long)m * N + (n0 + l15)] = v;
  }
}

// ---------------- RPN heads: per-(pixel,anchor) score + 4 deltas ----------------
__global__ void k_rpn_heads(const float* __restrict__ h, const float* __restrict__ wc,
                            const float* __restrict__ bc, const float* __restrict__ wb,
                            const float* __restrict__ bbv, float* __restrict__ sc,
                            float* __restrict__ dl, int HW)
{
  long i = (long)blockIdx.x * 256 + threadIdx.x;
  if (i >= (long)HW * 15) return;
  int p = (int)(i / 15), a = (int)(i - (long)p * 15);
  const float* hr = h + (long)p * 512;
  float c0 = bc[2 * a], c1 = bc[2 * a + 1];
  const float* w0 = wc + (long)(2 * a) * 512;
  const float* w1 = w0 + 512;
  for (int k = 0; k < 512; ++k) { float hv = hr[k]; c0 += hv * w0[k]; c1 += hv * w1[k]; }
  sc[i] = 1.0f / (1.0f + expf(c0 - c1));
#pragma unroll
  for (int d = 0; d < 4; ++d) {
    const float* wd = wb + (long)(4 * a + d) * 512;
    float s = bbv[4 * a + d];
    for (int k = 0; k < 512; ++k) s += hr[k] * wd[k];
    dl[i * 4 + d] = s;
  }
}

// ---------------- top-50 selection (desc, low-index ties) + anchor decode ----------------
__global__ void k_topk_decode(float* __restrict__ sc, const float* __restrict__ dl,
                              float* __restrict__ boxes, int A, int W, float s)
{
  __shared__ float sv[256];
  __shared__ int si[256];
  int tid = threadIdx.x;
  for (int it = 0; it < 50; ++it) {
    float bv = -3.0e38f; int bi = 0x7fffffff;
    for (int i = tid; i < A; i += 256) {
      float v = sc[i];
      if (v > bv || (v == bv && i < bi)) { bv = v; bi = i; }
    }
    sv[tid] = bv; si[tid] = bi;
    __syncthreads();
    for (int off = 128; off > 0; off >>= 1) {
      if (tid < off) {
        float v2 = sv[tid + off]; int i2 = si[tid + off];
        if (v2 > sv[tid] || (v2 == sv[tid] && i2 < si[tid])) { sv[tid] = v2; si[tid] = i2; }
      }
      __syncthreads();
    }
    if (tid == 0) {
      int idx = si[0];
      sc[idx] = -3.4e38f;
      int p = idx / 15, a = idx - p * 15;
      int yy = p / W, xx = p - yy * W;
      float w = 0.f, hh = 0.f, ph = 0.f;
      if (a <= 10) {   // replicate reference's overlapping base-row writes
        int ri = (a < 3) ? 0 : ((a < 6) ? 1 : 2);
        int sj = (a < 3) ? a : ((a < 6) ? a - 3 : a - 6);
        const float scl[5] = {4.f, 8.f, 16.f, 32.f, 64.f};
        const float sr[3] = {0.70710678118f, 1.0f, 1.41421356237f};
        w = s * scl[sj] * sr[ri];
        hh = s * scl[sj] / sr[ri];
        ph = 0.5f * s;
      }
      float d0 = dl[(long)idx * 4 + 0], d1 = dl[(long)idx * 4 + 1];
      float d2 = dl[(long)idx * 4 + 2], d3 = dl[(long)idx * 4 + 3];
      float cx = xx * s + ph, cy = yy * s + ph;
      float gx = cx + w * d0, gy = cy + hh * d1;
      float gw = w * expf(d2), gh = hh * expf(d3);
      float Wf = (float)W;
      float x1 = fminf(fmaxf(gx - 0.5f * gw, 0.0f), Wf);
      float y1 = fminf(fmaxf(gy - 0.5f * gh, 0.0f), Wf);
      boxes[it * 4 + 0] = x1; boxes[it * 4 + 1] = y1;
      boxes[it * 4 + 2] = x1 + gw; boxes[it * 4 + 3] = y1 + gh;
    }
    __syncthreads();
  }
}

// ---------------- ROI align ----------------
__device__ __forceinline__ float bilin(const float* __restrict__ ch, int H, int W, float yc, float xc) {
  float yf = floorf(yc), xf = floorf(xc);
  float fy = yc - yf, fx = xc - xf;
  int y0 = (int)yf, x0 = (int)xf;
  int y0c = min(max(y0, 0), H - 1);
  int y1c = min(max(y0 + 1, 0), H - 1);
  int x0c = min(max(x0, 0), W - 1);
  int x1c = min(max(x0 + 1, 0), W - 1);
  float v00 = ch[(long)y0c * W + x0c], v01 = ch[(long)y0c * W + x1c];
  float v10 = ch[(long)y1c * W + x0c], v11 = ch[(long)y1c * W + x1c];
  return v00 * (1 - fy) * (1 - fx) + v01 * (1 - fy) * fx + v10 * fy * (1 - fx) + v11 * fy * fx;
}

__device__ __forceinline__ float roival(const float* __restrict__ feat, const float* __restrict__ bx,
                                        int H, int W, float invs, int outn, int oy, int ox, int c)
{
  float x1 = bx[0] * invs, y1 = bx[1] * invs;
  float rw = fmaxf(bx[2] * invs - x1, 1.0f);
  float rh = fmaxf(bx[3] * invs - y1, 1.0f);
  const float* ch = feat + (long)c * H * W;
  float sum = 0.f;
#pragma unroll
  for (int sy = 0; sy < 2; ++sy)
#pragma unroll
    for (int sx = 0; sx < 2; ++sx) {
      int ky = oy * 2 + sy, kx = ox * 2 + sx;
      float py = (float)(ky >> 1) + (float)(ky & 1) * 0.5f + 0.25f;
      float px = (float)(kx >> 1) + (float)(kx & 1) * 0.5f + 0.25f;
      float yc = y1 + py * (rh / (float)outn);
      float xc = x1 + px * (rw / (float)outn);
      sum += bilin(ch, H, W, yc, xc);
    }
  return sum * 0.25f;
}

__global__ void k_roi7(const float* __restrict__ feat, const float* __restrict__ boxes,
                       bf16_t* __restrict__ p7bf, int H, int W, float invs, int lvl)
{
  long i = (long)blockIdx.x * 256 + threadIdx.x;
  if (i >= 50L * 256 * 49) return;
  int ox = (int)(i % 7);
  long r = i / 7;
  int oy = (int)(r % 7); r /= 7;
  int c = (int)(r % 256);
  int roi = (int)(r / 256);
  float v = roival(feat, boxes + roi * 4, H, W, invs, 7, oy, ox, c);
  p7bf[(long)(lvl * 50 + roi) * 12544 + (long)c * 49 + oy * 7 + ox] = (bf16_t)v;
}

__global__ void k_roi14(const float* __restrict__ feat, const float* __restrict__ boxes,
                        bf16_t* __restrict__ mbuf, int H, int W, float invs)
{
  long i = (long)blockIdx.x * 256 + threadIdx.x;
  if (i >= 50L * 256 * 196) return;
  int c = (int)(i % 256);
  long r = i / 256;
  int ox = (int)(r % 14); r /= 14;
  int oy = (int)(r % 14);
  int roi = (int)(r / 14);
  float v = roival(feat, boxes + roi * 4, H, W, invs, 14, oy, ox, c);
  mbuf[(((long)roi * 16 + oy + 1) * 16 + (ox + 1)) * 256 + c] = (bf16_t)v;
}

// ---------------- BN helpers ----------------
__global__ void k_bn_bf(const float* __restrict__ in, const float* __restrict__ g,
                        const float* __restrict__ be, bf16_t* __restrict__ out, int N, long total)
{
  long i = (long)blockIdx.x * 256 + threadIdx.x;
  if (i >= total) return;
  int n = (int)(i % N);
  out[i] = (bf16_t)(in[i] * (g[n] * rsqrtf(1.001f)) + be[n]);
}

__global__ void k_bn_f32(const float* __restrict__ in, const float* __restrict__ g,
                         const float* __restrict__ be, float* __restrict__ out, int N, long total, int relu)
{
  long i = (long)blockIdx.x * 256 + threadIdx.x;
  if (i >= total) return;
  int n = (int)(i % N);
  float v = in[i] * (g[n] * rsqrtf(1.001f)) + be[n];
  if (relu) v = fmaxf(v, 0.0f);
  out[i] = v;
}

// mconv [50][196][256] f32 -> bn+relu -> padded NHWC bf16 (borders zero)
__global__ void k_repad(const float* __restrict__ mconv, const float* __restrict__ g,
                        const float* __restrict__ be, bf16_t* __restrict__ mbuf)
{
  long i = (long)blockIdx.x * 256 + threadIdx.x;
  if (i >= 50L * 16 * 16 * 256) return;
  int c = (int)(i % 256);
  long r = i / 256;
  int px = (int)(r % 16); r /= 16;
  int py = (int)(r % 16);
  int roi = (int)(r / 16);
  float v = 0.0f;
  if (py >= 1 && py <= 14 && px >= 1 && px <= 14) {
    float t = mconv[((long)roi * 196 + (py - 1) * 14 + (px - 1)) * 256 + c];
    v = fmaxf(t * (g[c] * rsqrtf(1.001f)) + be[c], 0.0f);
  }
  mbuf[i] = (bf16_t)v;
}

__global__ void k_bnrelu_inplace(float* __restrict__ x, const float* __restrict__ g,
                                 const float* __restrict__ be, long total)
{
  long i = (long)blockIdx.x * 256 + threadIdx.x;
  if (i >= total) return;
  int c = (int)(i % 256);
  x[i] = fmaxf(x[i] * (g[c] * rsqrtf(1.001f)) + be[c], 0.0f);
}

// ---------------- deconv 2x2 stride2 + relu ----------------
__global__ void k_deconv(const float* __restrict__ mconv, const float* __restrict__ wdt,
                         const float* __restrict__ db, float* __restrict__ mdec)
{
  long i = (long)blockIdx.x * 256 + threadIdx.x;
  if (i >= 50L * 784 * 256) return;
  int o = (int)(i % 256);
  long r = i / 256;
  int pix = (int)(r % 784);
  int roi = (int)(r / 784);
  int ox = pix % 28, oy = pix / 28;
  int sy = oy >> 1, sx = ox >> 1;
  int ab = (oy & 1) * 2 + (ox & 1);
  const float* in = mconv + ((long)roi * 196 + sy * 14 + sx) * 256;
  const float* wr = wdt + ((long)ab * 256 + o) * 256;
  float s = db[o];
  for (int k = 0; k < 256; ++k) s += in[k] * wr[k];
  mdec[i] = fmaxf(s, 0.0f);
}

// ---------------- final 1x1 conv + sigmoid -> d_out mask region ----------------
__global__ void k_maskfinal(const float* __restrict__ mdec, const float* __restrict__ fw,
                            const float* __restrict__ fb, float* __restrict__ out, int lvl)
{
  long i = (long)blockIdx.x * 256 + threadIdx.x;
  if (i >= 50L * 784 * 21) return;
  int cls = (int)(i % 21);
  long r = i / 21;
  int pix = (int)(r % 784);
  int roi = (int)(r / 784);
  const float* in = mdec + ((long)roi * 784 + pix) * 256;
  const float* wr = fw + (long)cls * 256;
  float s = fb[cls];
  for (int k = 0; k < 256; ++k) s += in[k] * wr[k];
  out[((long)(lvl * 50 + roi) * 21 + cls) * 784 + pix] = 1.0f / (1.0f + expf(-s));
}

// ---------------- FC: out[m][n] = h[m] . w[n] + b[n] ----------------
__global__ void k_fc(const float* __restrict__ h, const float* __restrict__ w,
                     const float* __restrict__ b, float* __restrict__ out, int M, int N, int K)
{
  long i = (long)blockIdx.x * 256 + threadIdx.x;
  if (i >= (long)M * N) return;
  int n = (int)(i % N), m = (int)(i / N);
  const float* hr = h + (long)m * K;
  const float* wr = w + (long)n * K;
  float s = b[n];
  for (int k = 0; k < K; ++k) s += hr[k] * wr[k];
  out[i] = s;
}

// ================= host orchestration =================
static const long PSZ[26] = {1179648, 512, 15360, 30, 30720, 60, 12845056, 1024, 1024, 1024,
                             1048576, 1024, 1024, 1024, 21504, 21, 86016, 84, 2359296, 1024,
                             1024, 1024, 262144, 256, 5376, 21};
static const int ALPHA[26] = {25, 22, 24, 21, 23, 20, 5, 4, 10, 2, 7, 6, 11, 3, 9, 8, 1, 0,
                              19, 12, 18, 13, 15, 14, 17, 16};
enum { P_RPN_WS = 0, P_RPN_BS, P_RPN_WC, P_RPN_BC, P_RPN_WB, P_RPN_BB,
       P_CLS_C1W, P_CLS_C1B, P_CLS_G1, P_CLS_BE1, P_CLS_C2W, P_CLS_C2B,
       P_CLS_G2, P_CLS_BE2, P_CLS_FCW, P_CLS_FCB, P_CLS_BBW, P_CLS_BBB,
       P_MSK_W, P_MSK_B, P_MSK_G, P_MSK_BE, P_MSK_DW, P_MSK_DB, P_MSK_FW, P_MSK_FB };

extern "C" void kernel_launch(void* const* d_in, const int* in_sizes, int n_in,
                              void* d_out, int out_size, void* d_ws, size_t ws_size,
                              hipStream_t stream)
{
  (void)out_size; (void)ws_size;
  const float* feat[5];
  for (int i = 0; i < 5; ++i) feat[i] = (const float*)d_in[i];

  const float* prm[26];
  if (n_in >= 31) {
    bool alpha = (in_sizes[5] != 1179648);
    for (int k = 0; k < 26; ++k)
      prm[k] = (const float*)d_in[5 + (alpha ? ALPHA[k] : k)];
  } else {
    const float* base = (const float*)d_in[5];
    long off = 0;
    for (int k = 0; k < 26; ++k) { prm[k] = base + off; off += PSZ[k]; }
  }

  unsigned char* W8 = (unsigned char*)d_ws;
  size_t off = 0;
  auto alloc = [&](size_t bytes) { size_t r = off; off = (off + bytes + 255) & ~(size_t)255; return r; };
  bf16_t* wrpn  = (bf16_t*)(W8 + alloc(512L * 9 * 256 * 2));
  bf16_t* wcls1 = (bf16_t*)(W8 + alloc(1024L * 12544 * 2));
  bf16_t* wcls2 = (bf16_t*)(W8 + alloc(1024L * 1024 * 2));
  bf16_t* wmsk  = (bf16_t*)(W8 + alloc(1024L * 9 * 256 * 2));
  float*  wdec  = (float*)(W8 + alloc(4L * 256 * 256 * 4));
  bf16_t* zpad  = (bf16_t*)(W8 + alloc(512));                 // zero page for invalid conv rows
  bf16_t* padf  = (bf16_t*)(W8 + alloc(130L * 130 * 256 * 2));
  float*  hbuf  = (float*)(W8 + alloc(16384L * 512 * 4));
  float*  score = (float*)(W8 + alloc(16384L * 15 * 4));
  float*  delta = (float*)(W8 + alloc(16384L * 15 * 16));
  float*  boxes = (float*)(W8 + alloc(50L * 4 * 4));
  bf16_t* p7bf  = (bf16_t*)(W8 + alloc(256L * 12544 * 2));
  float*  c1out = (float*)(W8 + alloc(256L * 1024 * 4));
  bf16_t* bn1bf = (bf16_t*)(W8 + alloc(256L * 1024 * 2));
  float*  c2out = (float*)(W8 + alloc(256L * 1024 * 4));
  float*  hcls  = (float*)(W8 + alloc(256L * 1024 * 4));
  bf16_t* mbufA = (bf16_t*)(W8 + alloc(50L * 16 * 16 * 256 * 2));
  bf16_t* mbufB = (bf16_t*)(W8 + alloc(50L * 16 * 16 * 256 * 2));
  float*  mconv = (float*)(W8 + alloc(50L * 196 * 256 * 4));
  float*  mdec  = (float*)(W8 + alloc(50L * 784 * 256 * 4));

  float* outF = (float*)d_out;
  float* out_logits = outF;          // [2][250][21]
  float* out_bb = outF + 10500;      // [2][250][84]
  float* out_mask = outF + 52500;    // [2][250][21][784]

  // ---- weight prep + zero page ----
  k_zero<<<1, 256, 0, stream>>>((unsigned int*)zpad, 128);
  k_cvt_w3<<<CDIV(512L * 9 * 256, 256), 256, 0, stream>>>(prm[P_RPN_WS], wrpn, 512L * 9 * 256, 256);
  k_cvt_w3<<<CDIV(1024L * 9 * 256, 256), 256, 0, stream>>>(prm[P_MSK_W], wmsk, 1024L * 9 * 256, 256);
  k_cvt<<<CDIV(1024L * 12544, 256), 256, 0, stream>>>(prm[P_CLS_C1W], wcls1, 1024L * 12544);
  k_cvt<<<CDIV(1024L * 1024, 256), 256, 0, stream>>>(prm[P_CLS_C2W], wcls2, 1024L * 1024);
  k_cvt_wdec<<<CDIV(4L * 256 * 256, 256), 256, 0, stream>>>(prm[P_MSK_DW], wdec, 4L * 256 * 256);

  static const int HS[5] = {128, 64, 32, 16, 8};
  static const int STRD[5] = {4, 8, 16, 32, 64};

  for (int img = 0; img < 2; ++img) {
    k_zero<<<CDIV(256L * 12544 * 2 / 4, 256), 256, 0, stream>>>((unsigned int*)p7bf, 256L * 12544 * 2 / 4);
    for (int lvl = 0; lvl < 5; ++lvl) {
      int Hh = HS[lvl], Ww = Hh, HW = Hh * Ww;
      float s = (float)STRD[lvl];
      float invs = 1.0f / s;
      const float* fimg = feat[lvl] + (long)img * 256 * HW;

      long padWords = (long)(Hh + 2) * (Ww + 2) * 256 * 2 / 4;
      k_zero<<<CDIV(padWords, 256), 256, 0, stream>>>((unsigned int*)padf, padWords);
      k_padfeat<<<CDIV((long)HW * 256, 256), 256, 0, stream>>>(fimg, padf, Hh, Ww);

      dim3 gC(HW / 16, 512 / 16, 1);
      k_conv3_wmma<<<gC, 32, 0, stream>>>(padf, wrpn, prm[P_RPN_BS], zpad, hbuf,
                                          Ww, Ww + 2, 256, 512, HW, 1, 0L, 0L);
      k_rpn_heads<<<CDIV((long)HW * 15, 256), 256, 0, stream>>>(
          hbuf, prm[P_RPN_WC], prm[P_RPN_BC], prm[P_RPN_WB], prm[P_RPN_BB], score, delta, HW);
      k_topk_decode<<<1, 256, 0, stream>>>(score, delta, boxes, HW * 15, Ww, s);

      k_roi7<<<CDIV(50L * 256 * 49, 256), 256, 0, stream>>>(fimg, boxes, p7bf, Hh, Ww, invs, lvl);
      k_zero<<<CDIV(50L * 16 * 16 * 256 * 2 / 4, 256), 256, 0, stream>>>(
          (unsigned int*)mbufA, 50L * 16 * 16 * 256 * 2 / 4);
      k_roi14<<<CDIV(50L * 256 * 196, 256), 256, 0, stream>>>(fimg, boxes, mbufA, Hh, Ww, invs);

      // mask head for this 50-ROI chunk
      bf16_t* cur = mbufA;
      bf16_t* nxt = mbufB;
      for (int l = 0; l < 4; ++l) {
        dim3 gm(CDIV(196, 16), 256 / 16, 50);
        k_conv3_wmma<<<gm, 32, 0, stream>>>(cur, wmsk + (long)l * 256 * 9 * 256,
                                            prm[P_MSK_B] + l * 256, zpad, mconv,
                                            14, 16, 256, 256, 196, 0, 16L * 16 * 256, 196L * 256);
        if (l < 3) {
          k_repad<<<CDIV(50L * 16 * 16 * 256, 256), 256, 0, stream>>>(
              mconv, prm[P_MSK_G] + l * 256, prm[P_MSK_BE] + l * 256, nxt);
          bf16_t* t = cur; cur = nxt; nxt = t;
        } else {
          k_bnrelu_inplace<<<CDIV(50L * 196 * 256, 256), 256, 0, stream>>>(
              mconv, prm[P_MSK_G] + l * 256, prm[P_MSK_BE] + l * 256, 50L * 196 * 256);
        }
      }
      k_deconv<<<CDIV(50L * 784 * 256, 256), 256, 0, stream>>>(mconv, wdec, prm[P_MSK_DB], mdec);
      k_maskfinal<<<CDIV(50L * 784 * 21, 256), 256, 0, stream>>>(
          mdec, prm[P_MSK_FW], prm[P_MSK_FB], out_mask + (long)img * 250 * 21 * 784, lvl);
    }
    // classifier over all 250 ROIs (M padded to 256)
    dim3 g1(16, 64, 1);
    k_gemm_wmma<<<g1, 32, 0, stream>>>(p7bf, wcls1, prm[P_CLS_C1B], c1out, 12544, 1024, 0);
    k_bn_bf<<<CDIV(256L * 1024, 256), 256, 0, stream>>>(c1out, prm[P_CLS_G1], prm[P_CLS_BE1],
                                                        bn1bf, 1024, 256L * 1024);
    k_gemm_wmma<<<g1, 32, 0, stream>>>(bn1bf, wcls2, prm[P_CLS_C2B], c2out, 1024, 1024, 0);
    k_bn_f32<<<CDIV(256L * 1024, 256), 256, 0, stream>>>(c2out, prm[P_CLS_G2], prm[P_CLS_BE2],
                                                         hcls, 1024, 256L * 1024, 1);
    k_fc<<<CDIV(250L * 21, 256), 256, 0, stream>>>(hcls, prm[P_CLS_FCW], prm[P_CLS_FCB],
                                                   out_logits + (long)img * 250 * 21, 250, 21, 1024);
    k_fc<<<CDIV(250L * 84, 256), 256, 0, stream>>>(hcls, prm[P_CLS_BBW], prm[P_CLS_BBB],
                                                   out_bb + (long)img * 250 * 84, 250, 84, 1024);
  }
}
